// DyadicMambaAdapter_7249904795699
// MI455X (gfx1250) — matrix operand order repaired and checked by reading, not compiled, not run on hardware
//
#include <hip/hip_runtime.h>

// ---------------------------------------------------------------------------
// DyadicMambaAdapter for gfx1250 (MI455X, wave32, WMMA)
//
// Shapes: B=2, L=1024, D_MODEL=1024, D_INNER=2048, N_STATE=16, DT_RANK=64
// All GEMMs run as bf16 WMMA (v_wmma_f32_16x16x32_bf16) with f32 accumulate.
// Tile staging uses GLOBAL_LOAD_ASYNC_TO_LDS_B128 (ASYNCcnt) with LDS double
// buffering so the DMA of tile t+1 overlaps the WMMAs of tile t.
// The selective scan is fused (dA -> quantize -> scan -> h.C reduction) so the
// 256MB (B,L,D,N) intermediates never touch HBM.
// ---------------------------------------------------------------------------

#define B_SZ     2
#define L_SEQ    1024
#define D_MODEL  1024
#define D_INNER  2048
#define N_STATE  16
#define DT_RANK  64
#define MROWS    (B_SZ * L_SEQ)        // 2048

typedef __attribute__((ext_vector_type(16))) __bf16 v16bf;
typedef __attribute__((ext_vector_type(8)))  float  v8f;
typedef __attribute__((ext_vector_type(4)))  int    v4i;

union Frag16 { v16bf v; unsigned int u[8]; };
union FragC  { v8f  v; float f[8]; };

#define LDS_AS __attribute__((address_space(3)))
#define GLB_AS __attribute__((address_space(1)))

#if defined(__AMDGCN__) && __has_builtin(__builtin_amdgcn_global_load_async_to_lds_b128)
#define HAVE_ASYNC_COPY 1
#else
#define HAVE_ASYNC_COPY 0
#endif

__device__ __forceinline__ void wait_async0() {
#if HAVE_ASYNC_COPY
#if __has_builtin(__builtin_amdgcn_s_wait_asynccnt)
  __builtin_amdgcn_s_wait_asynccnt(0);
#else
  asm volatile("s_wait_asynccnt 0x0" ::: "memory");
#endif
#endif
}

__device__ inline unsigned short f2bf(float f) {
  unsigned int u = __float_as_uint(f);
  u += 0x7FFFu + ((u >> 16) & 1u);          // round-to-nearest-even
  return (unsigned short)(u >> 16);
}

__device__ inline float sigmoidf_fast(float x) {
  return 1.0f / (1.0f + __expf(-x));
}

// ------------------------------ converts -----------------------------------

__global__ void k_f32_to_bf16(const float* __restrict__ in,
                              unsigned short* __restrict__ out, int n) {
  int i = blockIdx.x * blockDim.x + threadIdx.x;
  if (i < n) out[i] = f2bf(in[i]);
}

// take first `outcols` of each `incols`-wide row
__global__ void k_slice_to_bf16(const float* __restrict__ in,
                                unsigned short* __restrict__ out,
                                int rows, int incols, int outcols) {
  int i = blockIdx.x * blockDim.x + threadIdx.x;
  if (i >= rows * outcols) return;
  int r = i / outcols, c = i - r * outcols;
  out[i] = f2bf(in[(size_t)r * incols + c]);
}

// ------------------------------ WMMA GEMM ----------------------------------
// C[M,N] = A[M,K](bf16,row-major) * Bw[N,K](bf16,row-major, i.e. X @ W^T)
// mode: 0 = plain, 1 = +bias, 2 = +bias then softplus
#define BM 128
#define BN 64
#define BK 32

// Stage one (A,B) tile pair into LDS buffer `la`/`lb` for k-slab k0.
__device__ __forceinline__
void stage_tiles(const unsigned short* __restrict__ A,
                 const unsigned short* __restrict__ Bw,
                 unsigned int (*la)[BK / 2], unsigned int (*lb)[BK / 2],
                 int K, int N, int bm, int bn, int k0, int tid) {
  // ---- A tile: 128x32 bf16, 2 x 16B per thread ----
#pragma unroll
  for (int it = 0; it < 2; ++it) {
    int e   = (tid + it * 256) * 8;
    int row = e >> 5, col = e & 31;
    const unsigned short* src = A + (size_t)(bm + row) * K + k0 + col;
    unsigned int* dst = &la[row][col >> 1];
#if HAVE_ASYNC_COPY
    __builtin_amdgcn_global_load_async_to_lds_b128(
        (GLB_AS v4i*)src, (LDS_AS v4i*)dst, 0, 0);
#else
    *(uint4*)dst = *(const uint4*)src;
#endif
  }
  // ---- B tile: 64x32 bf16, 16B per thread, N-guarded (N=96 case) ----
  {
    int e   = tid * 8;
    int row = e >> 5, col = e & 31;
    unsigned int* dst = &lb[row][col >> 1];
    if (bn + row < N) {
      const unsigned short* src = Bw + (size_t)(bn + row) * K + k0 + col;
#if HAVE_ASYNC_COPY
      __builtin_amdgcn_global_load_async_to_lds_b128(
          (GLB_AS v4i*)src, (LDS_AS v4i*)dst, 0, 0);
#else
      *(uint4*)dst = *(const uint4*)src;
#endif
    } else {
      uint4 z; z.x = z.y = z.z = z.w = 0u;
      *(uint4*)dst = z;
    }
  }
}

__global__ __launch_bounds__(256)
void k_gemm_bf16_wmma(const unsigned short* __restrict__ A,
                      const unsigned short* __restrict__ Bw,
                      const float* __restrict__ bias,
                      float* __restrict__ C,
                      int M, int N, int K, int mode) {
  __shared__ unsigned int lds_a[2][BM][BK / 2];   // double buffered, 2 bf16/dword
  __shared__ unsigned int lds_b[2][BN][BK / 2];

  const int tid  = threadIdx.x;
  const int bm   = blockIdx.y * BM;
  const int bn   = blockIdx.x * BN;
  const int wave = tid >> 5;
  const int lane = tid & 31;
  const int half = lane >> 4;        // wave32: lanes 0-15 / 16-31
  const int l16  = lane & 15;
  const int wm   = (wave & 3) * 32;  // 4 waves along M
  const int wn   = (wave >> 2) * 32; // 2 waves along N

  FragC acc[2][2];
#pragma unroll
  for (int i = 0; i < 2; ++i)
#pragma unroll
    for (int j = 0; j < 2; ++j)
#pragma unroll
      for (int v = 0; v < 8; ++v) acc[i][j].f[v] = 0.f;

  const int T = K / BK;
  stage_tiles(A, Bw, lds_a[0], lds_b[0], K, N, bm, bn, 0, tid);

  for (int t = 0; t < T; ++t) {
    const int buf = t & 1;
    wait_async0();          // own async copies for tile t landed in LDS
    __syncthreads();        // all waves' copies visible; prev reads of buf done

    // ---- gather fragments per documented 16-bit WMMA VGPR layouts ----
    Frag16 afrag[2], bfrag[2];
#pragma unroll
    for (int mt = 0; mt < 2; ++mt) {
      int m = wm + mt * 16 + l16;               // A: lane = row M
#pragma unroll
      for (int v = 0; v < 8; ++v) {
        // dword v holds K pair; K = 2*(v&3) + 8*half + 16*(v>>2)
        int kd = (v & 3) + 4 * half + 8 * (v >> 2);
        afrag[mt].u[v] = lds_a[buf][m][kd];
      }
    }
#pragma unroll
    for (int nt = 0; nt < 2; ++nt) {
      int n = wn + nt * 16 + l16;               // B: lane = col N
#pragma unroll
      for (int v = 0; v < 8; ++v)               // K pair = 2*(v&3) + 8*half + 16*(v>>2)
        bfrag[nt].u[v] = lds_b[buf][n][(v & 3) + 4 * half + 8 * (v >> 2)];
    }

    __syncthreads();        // everyone issued reads of buf; safe to refill other
    if (t + 1 < T)          // DMA of tile t+1 overlaps the WMMAs below
      stage_tiles(A, Bw, lds_a[buf ^ 1], lds_b[buf ^ 1],
                  K, N, bm, bn, (t + 1) * BK, tid);

#pragma unroll
    for (int mt = 0; mt < 2; ++mt)
#pragma unroll
      for (int nt = 0; nt < 2; ++nt)
        acc[mt][nt].v = __builtin_amdgcn_wmma_f32_16x16x32_bf16(
            false, afrag[mt].v, false, bfrag[nt].v,
            (short)0, acc[mt][nt].v, false, false);
  }

  // ---- epilogue: C/D layout lane=N, VGPR v -> M = v + 8*half ----
#pragma unroll
  for (int mt = 0; mt < 2; ++mt) {
#pragma unroll
    for (int nt = 0; nt < 2; ++nt) {
      int n = bn + wn + nt * 16 + l16;
      if (n >= N) continue;
      float bv = (mode >= 1) ? bias[n] : 0.f;
#pragma unroll
      for (int v = 0; v < 8; ++v) {
        int m = bm + wm + mt * 16 + v + 8 * half;
        float r = acc[mt][nt].f[v] + bv;
        if (mode == 2) r = (r > 20.f) ? r : log1pf(__expf(r));
        C[(size_t)m * N + n] = r;
      }
    }
  }
}

// ------------------- depthwise causal conv (K=4) + SiLU --------------------
__global__ void k_conv_silu(const float* __restrict__ proj,   // (M, 2*D_INNER), x = [:, :D_INNER]
                            const float* __restrict__ conv_w, // (D_INNER,1,4)
                            const float* __restrict__ conv_b,
                            float* __restrict__ xf,
                            unsigned short* __restrict__ xbf) {
  int idx = blockIdx.x * blockDim.x + threadIdx.x;   // over B*L*D_INNER
  if (idx >= B_SZ * L_SEQ * D_INNER) return;
  int d = idx & (D_INNER - 1);
  int l = (idx >> 11) & (L_SEQ - 1);
  int b = idx >> 21;
  float acc = conv_b[d];
#pragma unroll
  for (int j = 0; j < 4; ++j) {
    int ls = l - 3 + j;                                // left pad 3, causal
    if (ls >= 0)
      acc += proj[((size_t)(b * L_SEQ + ls)) * (2 * D_INNER) + d] * conv_w[d * 4 + j];
  }
  float s = acc * sigmoidf_fast(acc);                  // SiLU
  xf[idx]  = s;
  xbf[idx] = f2bf(s);
}

// ---------------- fused selective scan + quantized decay -------------------
// wave32: lanes 0-15 = the 16 SSM states of channel c0, lanes 16-31 = c1.
// h stays in a register for the whole L=1024 scan; y = sum_n h*C + x*D.
__global__ __launch_bounds__(256)
void k_scan(const float* __restrict__ dt,     // (M, D_INNER)
            const float* __restrict__ xf,     // (M, D_INNER)
            const float* __restrict__ xdbl,   // (M, 96): [64:80)=Bc, [80:96)=Cc
            const float* __restrict__ A_log,  // (D_INNER, 16)
            const float* __restrict__ Dvec,   // (D_INNER)
            float* __restrict__ ycore) {      // (M, D_INNER)
  const int tid  = threadIdx.x;
  const int n    = tid & 15;                            // state index
  const int chan = blockIdx.x * (blockDim.x >> 4) + (tid >> 4);
  const int b    = chan / D_INNER;
  const int d    = chan - b * D_INNER;

  const float a    = -__expf(A_log[d * N_STATE + n]);
  const float Dval = Dvec[d];
  float h = 0.f;

  for (int l = 0; l < L_SEQ; ++l) {
    const size_t row = (size_t)(b * L_SEQ + l);
    float dtv = dt[row * D_INNER + d];                  // broadcast across 16 lanes
    float xv  = xf[row * D_INNER + d];
    float Bv  = xdbl[row * 96 + DT_RANK + n];
    float Cv  = xdbl[row * 96 + DT_RANK + N_STATE + n];

    float dA    = __expf(dtv * a);
    float num   = fminf(fmaxf(floorf(dA * 32768.0f), 0.0f), 32767.0f);
    float decay = num * (1.0f / 32768.0f);              // dyadic fixed point
    float u     = xv * Bv * dtv;
    h = decay * h + u;

    float contrib = h * Cv;
#pragma unroll
    for (int off = 8; off; off >>= 1)                   // 16-lane butterfly
      contrib += __shfl_xor(contrib, off, 16);

    if (n == 0) ycore[row * D_INNER + d] = contrib + xv * Dval;
  }
}

// ------------------------- y * silu(z), cast bf16 --------------------------
__global__ void k_gate_silu(const float* __restrict__ ycore,
                            const float* __restrict__ proj,   // z = proj[:, D_INNER:]
                            unsigned short* __restrict__ ybf) {
  int idx = blockIdx.x * blockDim.x + threadIdx.x;      // over M*D_INNER
  if (idx >= MROWS * D_INNER) return;
  int m = idx >> 11, d = idx & (D_INNER - 1);
  float z = proj[(size_t)m * (2 * D_INNER) + D_INNER + d];
  ybf[idx] = f2bf(ycore[idx] * z * sigmoidf_fast(z));
}

// ---------------------------------------------------------------------------

extern "C" void kernel_launch(void* const* d_in, const int* in_sizes, int n_in,
                              void* d_out, int out_size, void* d_ws, size_t ws_size,
                              hipStream_t stream) {
  const float* hs        = (const float*)d_in[0];
  const float* in_proj_w = (const float*)d_in[1];
  const float* in_proj_b = (const float*)d_in[2];
  const float* conv_w    = (const float*)d_in[3];
  const float* conv_b    = (const float*)d_in[4];
  const float* x_proj_w  = (const float*)d_in[5];
  const float* dt_proj_w = (const float*)d_in[6];
  const float* dt_proj_b = (const float*)d_in[7];
  const float* out_proj_w= (const float*)d_in[8];
  const float* out_proj_b= (const float*)d_in[9];
  const float* A_log     = (const float*)d_in[10];
  const float* Dvec      = (const float*)d_in[11];
  float* outp            = (float*)d_out;

  // ---- workspace carve (256B aligned) ----
  char* ws = (char*)d_ws;
  size_t off = 0;
  auto alloc = [&](size_t bytes) -> void* {
    void* p = ws + off;
    off = (off + bytes + 255) & ~(size_t)255;
    return p;
  };
  unsigned short* hs_bf   = (unsigned short*)alloc((size_t)MROWS * D_MODEL * 2);
  unsigned short* win_bf  = (unsigned short*)alloc((size_t)2 * D_INNER * D_MODEL * 2);
  unsigned short* wx_bf   = (unsigned short*)alloc((size_t)96 * D_INNER * 2);
  unsigned short* wdt_bf  = (unsigned short*)alloc((size_t)D_INNER * DT_RANK * 2);
  unsigned short* wout_bf = (unsigned short*)alloc((size_t)D_MODEL * D_INNER * 2);
  float*          proj    = (float*)alloc((size_t)MROWS * 2 * D_INNER * 4);
  float*          xf      = (float*)alloc((size_t)MROWS * D_INNER * 4);
  unsigned short* x_bf    = (unsigned short*)alloc((size_t)MROWS * D_INNER * 2);
  float*          xdbl    = (float*)alloc((size_t)MROWS * 96 * 4);
  unsigned short* dtin_bf = (unsigned short*)alloc((size_t)MROWS * DT_RANK * 2);
  float*          dtbuf   = (float*)alloc((size_t)MROWS * D_INNER * 4);
  float*          ycore   = (float*)alloc((size_t)MROWS * D_INNER * 4);
  unsigned short* y_bf    = (unsigned short*)alloc((size_t)MROWS * D_INNER * 2);
  (void)ws_size; (void)in_sizes; (void)n_in; (void)out_size;

  const int TPB = 256;
  auto cvt = [&](const float* src, unsigned short* dst, int n) {
    k_f32_to_bf16<<<(n + TPB - 1) / TPB, TPB, 0, stream>>>(src, dst, n);
  };

  // 1) bf16 conversions
  cvt(hs,         hs_bf,   MROWS * D_MODEL);
  cvt(in_proj_w,  win_bf,  2 * D_INNER * D_MODEL);
  cvt(x_proj_w,   wx_bf,   96 * D_INNER);
  cvt(dt_proj_w,  wdt_bf,  D_INNER * DT_RANK);
  cvt(out_proj_w, wout_bf, D_MODEL * D_INNER);

  // 2) in_proj: (2048x1024) @ (4096x1024)^T + b -> proj (2048x4096)
  {
    dim3 grid((2 * D_INNER + BN - 1) / BN, MROWS / BM);
    k_gemm_bf16_wmma<<<grid, 256, 0, stream>>>(hs_bf, win_bf, in_proj_b, proj,
                                               MROWS, 2 * D_INNER, D_MODEL, 1);
  }

  // 3) depthwise causal conv + SiLU -> xf, x_bf
  k_conv_silu<<<(MROWS * D_INNER + TPB - 1) / TPB, TPB, 0, stream>>>(
      proj, conv_w, conv_b, xf, x_bf);

  // 4) x_proj: (2048x2048) @ (96x2048)^T -> xdbl (2048x96)
  {
    dim3 grid((96 + BN - 1) / BN, MROWS / BM);
    k_gemm_bf16_wmma<<<grid, 256, 0, stream>>>(x_bf, wx_bf, nullptr, xdbl,
                                               MROWS, 96, D_INNER, 0);
  }

  // 5) dt input slice -> bf16
  k_slice_to_bf16<<<(MROWS * DT_RANK + TPB - 1) / TPB, TPB, 0, stream>>>(
      xdbl, dtin_bf, MROWS, 96, DT_RANK);

  // 6) dt_proj: (2048x64) @ (2048x64)^T + b, softplus -> dtbuf (2048x2048)
  {
    dim3 grid((D_INNER + BN - 1) / BN, MROWS / BM);
    k_gemm_bf16_wmma<<<grid, 256, 0, stream>>>(dtin_bf, wdt_bf, dt_proj_b, dtbuf,
                                               MROWS, D_INNER, DT_RANK, 2);
  }

  // 7) fused scan: 4096 channels, 2 channels/wave, 16 channels/block
  {
    int chans  = B_SZ * D_INNER;
    int blocks = chans / (256 / 16);
    k_scan<<<blocks, 256, 0, stream>>>(dtbuf, xf, xdbl, A_log, Dvec, ycore);
  }

  // 8) gate with silu(z) -> bf16
  k_gate_silu<<<(MROWS * D_INNER + TPB - 1) / TPB, TPB, 0, stream>>>(
      ycore, proj, y_bf);

  // 9) out_proj: (2048x2048) @ (1024x2048)^T + b -> d_out (2048x1024)
  {
    dim3 grid((D_MODEL + BN - 1) / BN, MROWS / BM);
    k_gemm_bf16_wmma<<<grid, 256, 0, stream>>>(y_bf, wout_bf, out_proj_b, outp,
                                               MROWS, D_MODEL, D_INNER, 1);
  }
}
